// SelfAttention_69492570849585
// MI455X (gfx1250) — matrix operand order repaired and checked by reading, not compiled, not run on hardware
//
#include <hip/hip_runtime.h>

#define DEV_INLINE __device__ __forceinline__

typedef __attribute__((ext_vector_type(16))) _Float16 v16h;
typedef __attribute__((ext_vector_type(8)))  _Float16 v8h;
typedef __attribute__((ext_vector_type(8)))  float    v8f;
typedef __attribute__((ext_vector_type(4)))  unsigned u32x4;
typedef __attribute__((ext_vector_type(8)))  int      i32x8;
typedef __attribute__((ext_vector_type(4)))  int      i32x4;

static constexpr int BATCH = 16;
static constexpr int C = 256;
static constexpr int N = 1024;           // 32*32 tokens
static constexpr float EPS = 1e-6f;
static constexpr float SCALE = 0.0625f;  // C^-0.5 = 1/16

DEV_INLINE v8f wmma_f16(v16h a, v16h b, v8f c) {
  // D = A(16x32 f16) * B(32x16 f16) + C(16x16 f32)
  return __builtin_amdgcn_wmma_f32_16x16x32_f16(false, a, false, b,
                                                (short)0, c, false, false);
}

// A-fragment (16x32 f16). rowbase = &tile[m][k0]. Lane (m = lane&15, g = lane>>4)
// holds halves j -> K = g*16+j (contiguous, matches B fragments).
DEV_INLINE v16h load_a_frag(const _Float16* rowbase, int g) {
  v8h lo = *(const v8h*)(rowbase + g * 16);
  v8h hi = *(const v8h*)(rowbase + g * 16 + 8);
  v16h r;
#pragma unroll
  for (int i = 0; i < 8; ++i) { r[i] = lo[i]; r[i + 8] = hi[i]; }
  return r;
}

// Async global->LDS 16-byte copy, per lane (ASYNCcnt path, no VGPR data).
DEV_INLINE void async_copy_b128(unsigned lds_addr, const void* gaddr) {
  asm volatile("global_load_async_to_lds_b128 %0, %1, off"
               :: "v"(lds_addr), "v"(gaddr) : "memory");
}
DEV_INLINE void wait_async0() {
  asm volatile("s_wait_asynccnt 0x0" ::: "memory");
}

// Tensor Data Mover: 2-D tile load, 32 rows x 256 cols of f16 from a
// 1024x256 (row-pitch 256) tensor into LDS. D# layout per ISA 08 §8.3/8.4.
// This toolchain exposes the 6-arg builtin:
//   (u32x4 group0, i32x8 group1, i32x4 group2, i32x4 group3, i32x8, i32 cpol)
DEV_INLINE void tdm_load_k_tile(unsigned lds_addr, const _Float16* gsrc) {
#if __has_builtin(__builtin_amdgcn_tensor_load_to_lds)
  unsigned long long ga = (unsigned long long)(uintptr_t)gsrc;
  u32x4 g0;
  g0[0] = 1u;                                                  // count=1, user D#
  g0[1] = lds_addr;                                            // lds_addr (bytes)
  g0[2] = (unsigned)ga;                                        // global_addr[31:0]
  g0[3] = (unsigned)((ga >> 32) & 0x01FFFFFFu) | (2u << 30);   // addr[56:32], type=2
  i32x8 g1;
  g1[0] = (int)(1u << 16);     // wg_mask=0, data_size=1 (2-byte elements)
  g1[1] = (int)(256u << 16);   // tensor_dim0 = 256        (bits 79:48)
  g1[2] = (int)(1024u << 16);  // tensor_dim1 = 1024       (bits 111:80)
  g1[3] = (int)(256u << 16);   // tile_dim0 = 256          (bits 127:112)
  g1[4] = 32;                  // tile_dim1 = 32           (bits 143:128)
  g1[5] = 256;                 // tensor_dim0_stride = 256 (bits 207:160)
  g1[6] = 0;
  g1[7] = 0;
  i32x4 z4 = {0, 0, 0, 0};
  i32x8 z8 = {0, 0, 0, 0, 0, 0, 0, 0};
  __builtin_amdgcn_tensor_load_to_lds(g0, g1, z4, z4, z8, 0);
  __builtin_amdgcn_s_wait_tensorcnt(0);
#else
  (void)lds_addr; (void)gsrc;
#endif
}

// ---------------------------------------------------------------- weights f32->f16
__global__ void cvt_w_kernel(const float* __restrict__ wq, const float* __restrict__ wk,
                             const float* __restrict__ wv, const float* __restrict__ wp,
                             _Float16* __restrict__ dst) {
  int gidx = blockIdx.x * blockDim.x + threadIdx.x;  // 0 .. 4*C*C-1
  int mat = gidx >> 16;
  int off = gidx & 0xFFFF;
  const float* src = (mat == 0) ? wq : (mat == 1) ? wk : (mat == 2) ? wv : wp;
  dst[(size_t)mat * (C * C) + off] = (_Float16)src[off];
}

// ---------------------------------------------------------------- LayerNorm2d -> hn[B,N,C] f16
__global__ void ln_kernel(const float* __restrict__ x, const float* __restrict__ lnw,
                          const float* __restrict__ lnb, _Float16* __restrict__ hn) {
  int p = blockIdx.x * blockDim.x + threadIdx.x;  // pixel id, coalesced in n
  int b = p >> 10;
  int n = p & (N - 1);
  const float* xp = x + (size_t)b * C * N + n;
  float s = 0.f, ss = 0.f;
#pragma unroll 8
  for (int c = 0; c < C; ++c) {
    float v = xp[(size_t)c * N];
    s += v; ss += v * v;
  }
  float mean = s * (1.0f / C);
  float var  = ss * (1.0f / C) - mean * mean;
  float rstd = rsqrtf(var + EPS);
  _Float16* hp = hn + ((size_t)b * N + n) * C;
#pragma unroll 8
  for (int c = 0; c < C; ++c) {
    float v = (xp[(size_t)c * N] - mean) * rstd * lnw[c] + lnb[c];
    hp[c] = (_Float16)v;
  }
}

// ---------------------------------------------------------------- QKV projection GEMM
// y[n,o] = sum_c hn[b,n,c] * W[o,c] + bias[o]
// vt_out==0: store tokens-major dst[b,n,o] ; vt_out==1: store channel-major dst[b,o,n]
__global__ void __launch_bounds__(256) qkv_kernel(const _Float16* __restrict__ hn,
                                                  const _Float16* __restrict__ w16,
                                                  const float* __restrict__ bias,
                                                  _Float16* __restrict__ dst,
                                                  int vt_out) {
  __shared__ __align__(16) _Float16 ldsA[64 * C];  // 32 KB token tile
  const int tid = threadIdx.x;
  const int b  = blockIdx.y;
  const int n0 = blockIdx.x * 64;

  {  // async-stage 64x256 f16 tile (rows contiguous in [N,C] layout)
    const uint4* src = (const uint4*)(hn + ((size_t)b * N + n0) * C);
    const unsigned la = (unsigned)(uintptr_t)ldsA;
#pragma unroll
    for (int i = 0; i < 8; ++i)
      async_copy_b128(la + (unsigned)(tid + i * 256) * 16,
                      (const void*)(src + tid + i * 256));
    wait_async0();
  }
  __syncthreads();

  const int wave = tid >> 5, lane = tid & 31;
  const int wr = wave & 3;   // 16-token row group
  const int wc = wave >> 2;  // 128-wide output-column group
  const int m = lane & 15, g = lane >> 4;

  v8f acc[8] = {};
  const _Float16* arow = ldsA + (wr * 16 + m) * C;

#pragma unroll
  for (int kc = 0; kc < 8; ++kc) {           // K = C in chunks of 32
    v16h a = load_a_frag(arow + kc * 32, g);
#pragma unroll
    for (int t = 0; t < 8; ++t) {            // 8 x 16-wide output tiles
      // B fragment: B[k=c][n=o] = W[o][c] -> contiguous 32B per lane
      v16h bt = *(const v16h*)(w16 + (size_t)(wc * 128 + t * 16 + m) * C +
                               kc * 32 + g * 16);
      acc[t] = wmma_f16(a, bt, acc[t]);
    }
  }

#pragma unroll
  for (int t = 0; t < 8; ++t) {
    int col = wc * 128 + t * 16 + m;
    float bcol = bias[col];
#pragma unroll
    for (int i = 0; i < 8; ++i) {            // C/D layout: row = i + 8*g, col = m
      int row = n0 + wr * 16 + i + 8 * g;
      _Float16 v = (_Float16)(acc[t][i] + bcol);
      if (vt_out)
        dst[((size_t)b * C + col) * N + row] = v;   // VT[b, c, n]
      else
        dst[((size_t)b * N + row) * C + col] = v;   // [b, n, c]
    }
  }
}

// ---------------------------------------------------------------- flash attention
// AO[b,n,c] = softmax(Q K^T * scale) V
// Q/K tokens-major [B,N,C] f16 ; VT channel-major [B,C,N] f16
__global__ void __launch_bounds__(128) attn_kernel(const _Float16* __restrict__ Q,
                                                   const _Float16* __restrict__ K,
                                                   const _Float16* __restrict__ VT,
                                                   _Float16* __restrict__ AO) {
  __shared__ __align__(16) _Float16 ldsK[32 * C];      // 16 KB, TDM-staged
  __shared__ __align__(16) _Float16 ldsP[4][16 * 32];  // per-wave P relayout scratch

  const int tid = threadIdx.x;
  const int wave = tid >> 5, lane = tid & 31;
  const int m = lane & 15, g = lane >> 4;
  const int b  = blockIdx.y;
  const int q0 = blockIdx.x * 64 + wave * 16;  // 16 queries per wave

  // Q fragments resident in VGPRs: 8 chunks of 16x32
  const _Float16* qp = Q + ((size_t)b * N + q0) * C;
  v16h aQ[8];
#pragma unroll
  for (int kc = 0; kc < 8; ++kc) aQ[kc] = load_a_frag(qp + m * C + kc * 32, g);

  v8f O[16] = {};           // 16x256 f32 accumulator (C/D fragments)
  float mst[8], lst[8];     // per-row online softmax stats (row = i + 8*g)
#pragma unroll
  for (int i = 0; i < 8; ++i) { mst[i] = -1e30f; lst[i] = 0.f; }

  const _Float16* kbase = K + (size_t)b * N * C;
  const _Float16* vt = VT + (size_t)b * C * N;
  _Float16* pw = ldsP[wave];
  const unsigned lk = (unsigned)(uintptr_t)ldsK;  // LDS byte offset (low 32 bits)

  for (int kb = 0; kb < N; kb += 32) {
    __syncthreads();  // previous iteration done with ldsK
#if __has_builtin(__builtin_amdgcn_tensor_load_to_lds)
    if (wave == 0)    // one descriptor moves the whole 32x256 f16 tile
      tdm_load_k_tile(lk, kbase + (size_t)kb * C);
#else
    {  // fallback: per-lane async loads (ASYNCcnt path)
      const uint4* ksrc = (const uint4*)kbase;
      const int base = kb * (C / 8);
#pragma unroll
      for (int i = 0; i < 8; ++i)
        async_copy_b128(lk + (unsigned)(tid + i * 128) * 16,
                        (const void*)(ksrc + base + tid + i * 128));
      wait_async0();
    }
#endif
    if (kb + 32 < N)  // hint next K tile toward the caches
      __builtin_prefetch((const void*)(kbase + (size_t)(kb + 32) * C + tid * 16), 0, 0);
    __syncthreads();

    // S = Q * K^T for 32 keys -> two 16x16 f32 tiles
    v8f s0 = {}, s1 = {};
#pragma unroll
    for (int kc = 0; kc < 8; ++kc) {
      v16h bk0 = *(const v16h*)(ldsK + (size_t)m * C + kc * 32 + g * 16);
      v16h bk1 = *(const v16h*)(ldsK + (size_t)(16 + m) * C + kc * 32 + g * 16);
      s0 = wmma_f16(aQ[kc], bk0, s0);
      s1 = wmma_f16(aQ[kc], bk1, s1);
    }

    // online softmax; row stats reduced across the 16-lane half (cols live in lanes)
#pragma unroll
    for (int i = 0; i < 8; ++i) {
      float a0 = s0[i] * SCALE, a1 = s1[i] * SCALE;
      float t = fmaxf(a0, a1);
      t = fmaxf(t, __shfl_xor(t, 1, 32));
      t = fmaxf(t, __shfl_xor(t, 2, 32));
      t = fmaxf(t, __shfl_xor(t, 4, 32));
      t = fmaxf(t, __shfl_xor(t, 8, 32));
      float mn = fmaxf(mst[i], t);
      float p0 = __expf(a0 - mn), p1 = __expf(a1 - mn);
      float rs = p0 + p1;
      rs += __shfl_xor(rs, 1, 32);
      rs += __shfl_xor(rs, 2, 32);
      rs += __shfl_xor(rs, 4, 32);
      rs += __shfl_xor(rs, 8, 32);
      float corr = __expf(mst[i] - mn);
      lst[i] = lst[i] * corr + rs;
      mst[i] = mn;
#pragma unroll
      for (int t2 = 0; t2 < 16; ++t2) O[t2][i] *= corr;
      int row = i + 8 * g;                       // C/D -> (row, key) scatter to LDS
      pw[row * 32 + m]      = (_Float16)p0;
      pw[row * 32 + 16 + m] = (_Float16)p1;
    }

    // reload P in A-fragment layout (16x32)
    v16h aP = load_a_frag(pw + m * 32, g);

    // O += P * V : B[k=key][n=chan] = VT[chan][key] -> contiguous 32B per lane
#pragma unroll
    for (int t2 = 0; t2 < 16; ++t2) {
      v16h bv = *(const v16h*)(vt + (size_t)(t2 * 16 + m) * N + kb + g * 16);
      O[t2] = wmma_f16(aP, bv, O[t2]);
    }
  }

  // normalize rows and store f16 [B,N,C]
#pragma unroll
  for (int t2 = 0; t2 < 16; ++t2) {
#pragma unroll
    for (int i = 0; i < 8; ++i) {
      float o = O[t2][i] / lst[i];
      int row = q0 + i + 8 * g;
      AO[((size_t)b * N + row) * C + t2 * 16 + m] = (_Float16)o;
    }
  }
}

// ---------------------------------------------------------------- output projection + residual
// out[b,o,n] = sum_c AO[b,n,c]*Wp[o,c] + bp[o] + x[b,o,n]    (f32 out)
__global__ void __launch_bounds__(256) proj_kernel(const _Float16* __restrict__ ao,
                                                   const _Float16* __restrict__ wp16,
                                                   const float* __restrict__ bp,
                                                   const float* __restrict__ x,
                                                   float* __restrict__ out) {
  __shared__ __align__(16) _Float16 ldsA[64 * C];
  const int tid = threadIdx.x;
  const int b  = blockIdx.y;
  const int n0 = blockIdx.x * 64;
  {  // async-stage the attention-output tile
    const uint4* src = (const uint4*)(ao + ((size_t)b * N + n0) * C);
    const unsigned la = (unsigned)(uintptr_t)ldsA;
#pragma unroll
    for (int i = 0; i < 8; ++i)
      async_copy_b128(la + (unsigned)(tid + i * 256) * 16,
                      (const void*)(src + tid + i * 256));
    wait_async0();
  }
  __syncthreads();

  const int wave = tid >> 5, lane = tid & 31;
  const int wr = wave & 3, wc = wave >> 2;
  const int m = lane & 15, g = lane >> 4;

  v8f acc[8] = {};
  const _Float16* arow = ldsA + (wr * 16 + m) * C;
#pragma unroll
  for (int kc = 0; kc < 8; ++kc) {
    v16h a = load_a_frag(arow + kc * 32, g);
#pragma unroll
    for (int t = 0; t < 8; ++t) {
      v16h bt = *(const v16h*)(wp16 + (size_t)(wc * 128 + t * 16 + m) * C +
                               kc * 32 + g * 16);
      acc[t] = wmma_f16(a, bt, acc[t]);
    }
  }

#pragma unroll
  for (int t = 0; t < 8; ++t) {
    int col = wc * 128 + t * 16 + m;  // output channel o
    float bcol = bp[col];
#pragma unroll
    for (int i = 0; i < 8; ++i) {
      int row = n0 + wr * 16 + i + 8 * g;  // token n
      size_t idx = ((size_t)b * C + col) * N + row;
      out[idx] = acc[t][i] + bcol + x[idx];
    }
  }
}

// ----------------------------------------------------------------
extern "C" void kernel_launch(void* const* d_in, const int* in_sizes, int n_in,
                              void* d_out, int out_size, void* d_ws, size_t ws_size,
                              hipStream_t stream) {
  (void)in_sizes; (void)n_in; (void)out_size; (void)ws_size;
  const float* x   = (const float*)d_in[0];
  const float* lnw = (const float*)d_in[1];
  const float* lnb = (const float*)d_in[2];
  const float* wq  = (const float*)d_in[3];
  const float* bq  = (const float*)d_in[4];
  const float* wk  = (const float*)d_in[5];
  const float* bk  = (const float*)d_in[6];
  const float* wv  = (const float*)d_in[7];
  const float* bv  = (const float*)d_in[8];
  const float* wp  = (const float*)d_in[9];
  const float* bp  = (const float*)d_in[10];
  float* out = (float*)d_out;

  _Float16* ws = (_Float16*)d_ws;
  const size_t TOK = (size_t)BATCH * N * C;  // 4 Mi halfs per tensor
  _Float16* HN = ws;
  _Float16* Qm = ws + 1 * TOK;
  _Float16* Km = ws + 2 * TOK;
  _Float16* VT = ws + 3 * TOK;   // V stored transposed [B, C, N]
  _Float16* AO = ws + 4 * TOK;
  _Float16* W16 = ws + 5 * TOK;  // 4 x 65536 halfs
  _Float16* WQ = W16;
  _Float16* WK = W16 + 1 * C * C;
  _Float16* WV = W16 + 2 * C * C;
  _Float16* WP = W16 + 3 * C * C;

  cvt_w_kernel<<<dim3((4 * C * C) / 256), dim3(256), 0, stream>>>(wq, wk, wv, wp, W16);
  ln_kernel<<<dim3((BATCH * N) / 256), dim3(256), 0, stream>>>(x, lnw, lnb, HN);

  dim3 gg(N / 64, BATCH);
  qkv_kernel<<<gg, dim3(256), 0, stream>>>(HN, WQ, bq, Qm, 0);
  qkv_kernel<<<gg, dim3(256), 0, stream>>>(HN, WK, bk, Km, 0);
  qkv_kernel<<<gg, dim3(256), 0, stream>>>(HN, WV, bv, VT, 1);
  attn_kernel<<<gg, dim3(128), 0, stream>>>(Qm, Km, VT, AO);
  proj_kernel<<<gg, dim3(256), 0, stream>>>(AO, WP, bp, x, out);
}
